// TransformerDecoder_48584670052668
// MI455X (gfx1250) — matrix operand order repaired and checked by reading, not compiled, not run on hardware
//
#include <hip/hip_runtime.h>

// ---------------------------------------------------------------------------
// CDNA5 (gfx1250) transformer decoder
//   - all GEMM operands bf16 (A row-major [M][K], B pre-transposed [N][K])
//   - double-buffered global_load_async_to_lds_b128 staging (ASYNCcnt pipeline)
//   - v_wmma_f32_16x16x32_bf16 everywhere; fp32 residual stream
// ---------------------------------------------------------------------------

typedef __bf16 bf16;
typedef __attribute__((ext_vector_type(16))) __bf16 v16bf;
typedef __attribute__((ext_vector_type(8)))  float  v8f;
typedef __attribute__((ext_vector_type(8)))  unsigned int v8u;

static __device__ __forceinline__ unsigned short f32_to_bf16_bits(float f) {
  unsigned u = __builtin_bit_cast(unsigned, f);
  u += 0x7FFFu + ((u >> 16) & 1u);          // round-to-nearest-even
  return (unsigned short)(u >> 16);
}
static __device__ __forceinline__ unsigned pack_bf16x2(float lo, float hi) {
  return (unsigned)f32_to_bf16_bits(lo) | ((unsigned)f32_to_bf16_bits(hi) << 16);
}
static __device__ __forceinline__ v8f v8f_zero() {
  v8f z = {0.f,0.f,0.f,0.f,0.f,0.f,0.f,0.f};
  return z;
}
static __device__ __forceinline__ v8f wmma_bf16(v16bf a, v16bf b, v8f c) {
  return __builtin_amdgcn_wmma_f32_16x16x32_bf16(false, a, false, b, (short)0, c,
                                                 false, false);
}

// Raw 32-bit LDS offset of a __shared__ object (async-to-LDS VDST operand).
static __device__ __forceinline__ unsigned lds_off(const void* p) {
  return (unsigned)(size_t)(const __attribute__((address_space(3))) char*)p;
}
// Async 16B global->LDS copy (CDNA5, tracked with ASYNCcnt).
#define ASYNC_B128(ldsoff, gptr)                                              \
  asm volatile("global_load_async_to_lds_b128 %0, %1, off"                    \
               :: "v"(ldsoff), "v"((unsigned long long)(size_t)(gptr))        \
               : "memory")
#define WAIT_ASYNC(n) asm volatile("s_wait_asynccnt " #n ::: "memory")

// A-fragment (16x32 bf16, contraction window at kbase) from LDS tile [m][k].
static __device__ __forceinline__ v16bf lds_afrag(const unsigned short* tile,
                                                  int row_hw, int mbase, int kbase,
                                                  int lane) {
  int hf = lane >> 4;
  const unsigned short* rp = tile + (mbase + (lane & 15)) * row_hw + kbase + hf * 8;
  uint4 q0 = *(const uint4*)(rp);
  uint4 q1 = *(const uint4*)(rp + 16);
  v8u u = {q0.x, q0.y, q0.z, q0.w, q1.x, q1.y, q1.z, q1.w};
  return __builtin_bit_cast(v16bf, u);
}
// B-fragment (K=32 at kbase) from LDS tile stored transposed [n][k].
static __device__ __forceinline__ v16bf lds_bfrag(const unsigned short* tile,
                                                  int row_hw, int nbase, int kbase,
                                                  int lane) {
  int hf = lane >> 4;
  const unsigned short* rp = tile + (nbase + (lane & 15)) * row_hw + kbase + hf * 8;
  uint4 q0 = *(const uint4*)(rp);
  uint4 q1 = *(const uint4*)(rp + 16);
  v8u u = {q0.x, q0.y, q0.z, q0.w, q1.x, q1.y, q1.z, q1.w};
  return __builtin_bit_cast(v16bf, u);
}
// A-fragment straight from global bf16 row-major.
static __device__ __forceinline__ v16bf gld_afrag_bf16(const unsigned short* base,
                                                       int row_hw, int mbase,
                                                       int kbase, int lane) {
  int hf = lane >> 4;
  const unsigned short* rp = base + (size_t)(mbase + (lane & 15)) * row_hw + kbase + hf * 8;
  uint4 q0 = *(const uint4*)(rp);
  uint4 q1 = *(const uint4*)(rp + 16);
  v8u u = {q0.x, q0.y, q0.z, q0.w, q1.x, q1.y, q1.z, q1.w};
  return __builtin_bit_cast(v16bf, u);
}

// ---------------------------------------------------------------------------
// Transpose + convert: src f32 (K x N) -> dst bf16 (N x K).  Block 256 = 32x8.
// ---------------------------------------------------------------------------
__global__ __launch_bounds__(256) void transpose_f32_bf16(
    const float* __restrict__ src, unsigned short* __restrict__ dst,
    int Kdim, int Ndim, long long bsS, long long bsD) {
  __shared__ float t[32][33];
  src += (size_t)blockIdx.z * bsS;
  dst += (size_t)blockIdx.z * bsD;
  int k0 = blockIdx.y * 32, n0 = blockIdx.x * 32;
  int tx = threadIdx.x & 31, ty = threadIdx.x >> 5;
#pragma unroll
  for (int r = 0; r < 32; r += 8)
    t[ty + r][tx] = src[(size_t)(k0 + ty + r) * Ndim + n0 + tx];
  __syncthreads();
#pragma unroll
  for (int r = 0; r < 32; r += 8)
    dst[(size_t)(n0 + ty + r) * Kdim + k0 + tx] = f32_to_bf16_bits(t[tx][ty + r]);
}

// ---------------------------------------------------------------------------
// Per-head V re-tile: Vin bf16 (B*L x 512) -> VT bf16 [(b*8+h)*64+dh][1024].
// ---------------------------------------------------------------------------
__global__ __launch_bounds__(256) void transpose_v_heads(
    const unsigned short* __restrict__ Vin, unsigned short* __restrict__ VT) {
  __shared__ unsigned short t[32][68];
  int b = blockIdx.z, h = blockIdx.y, j0 = blockIdx.x * 32;
  const unsigned short* src = Vin + ((size_t)b * 1024 + j0) * 512 + h * 64;
  unsigned short* dst = VT + (size_t)(b * 8 + h) * 64 * 1024 + j0;
  int tid = threadIdx.x;
  {
    int j = tid >> 3, c = (tid & 7) * 8;
    uint4 q = *(const uint4*)(src + (size_t)j * 512 + c);
    t[j][c + 0] = (unsigned short)(q.x);  t[j][c + 1] = (unsigned short)(q.x >> 16);
    t[j][c + 2] = (unsigned short)(q.y);  t[j][c + 3] = (unsigned short)(q.y >> 16);
    t[j][c + 4] = (unsigned short)(q.z);  t[j][c + 5] = (unsigned short)(q.z >> 16);
    t[j][c + 6] = (unsigned short)(q.w);  t[j][c + 7] = (unsigned short)(q.w >> 16);
  }
  __syncthreads();
  {
    int dh = tid >> 2, jc = (tid & 3) * 8;
    unsigned pk[4];
#pragma unroll
    for (int u = 0; u < 4; ++u)
      pk[u] = (unsigned)t[jc + 2 * u][dh] | ((unsigned)t[jc + 2 * u + 1][dh] << 16);
    *(uint4*)(dst + (size_t)dh * 1024 + jc) = make_uint4(pk[0], pk[1], pk[2], pk[3]);
  }
}

// ---------------------------------------------------------------------------
// LayerNorm over last dim 512: f32 in -> bf16 out. One block per row.
// ---------------------------------------------------------------------------
__global__ __launch_bounds__(256) void layernorm_512_bf16(
    const float* __restrict__ X, const float* __restrict__ g,
    const float* __restrict__ b, unsigned short* __restrict__ Y) {
  __shared__ float red[256];
  int row = blockIdx.x;
  const float* x = X + (size_t)row * 512;
  unsigned* y = (unsigned*)(Y + (size_t)row * 512);
  int tid = threadIdx.x;
  float v0 = x[2 * tid], v1 = x[2 * tid + 1];
  red[tid] = v0 + v1;
  __syncthreads();
#pragma unroll
  for (int off = 128; off > 0; off >>= 1) {
    if (tid < off) red[tid] += red[tid + off];
    __syncthreads();
  }
  float mu = red[0] * (1.0f / 512.0f);
  __syncthreads();
  float d0 = v0 - mu, d1 = v1 - mu;
  red[tid] = d0 * d0 + d1 * d1;
  __syncthreads();
#pragma unroll
  for (int off = 128; off > 0; off >>= 1) {
    if (tid < off) red[tid] += red[tid + off];
    __syncthreads();
  }
  float rstd = rsqrtf(red[0] * (1.0f / 512.0f) + 1e-5f);
  float o0 = d0 * rstd * g[2 * tid] + b[2 * tid];
  float o1 = d1 * rstd * g[2 * tid + 1] + b[2 * tid + 1];
  y[tid] = pack_bf16x2(o0, o1);
}

// ---------------------------------------------------------------------------
// GEMM: C[m,n] = sum_k A[m][k] * BT[n][k]
// Block tile 128x128, K-step 32, 2-deep async pipeline, 8 waves x (32x64).
// Compile-time epilogue: BN (bias over n), BM (bias over m), GELU, RES, B16.
// ---------------------------------------------------------------------------
#define GEMM_ISSUE(kt, buf)                                                   \
  do {                                                                        \
    const unsigned short* _ag = agp + (kt) * 32;                              \
    const unsigned short* _bg = bgp + (kt) * 32;                              \
    unsigned _ao = aBase + (unsigned)(buf) * 8192u;                           \
    unsigned _bo = bBase + (unsigned)(buf) * 8192u;                           \
    ASYNC_B128(_ao, _ag);                                                     \
    ASYNC_B128(_ao + 16, _ag + 8);                                            \
    ASYNC_B128(_bo, _bg);                                                     \
    ASYNC_B128(_bo + 16, _bg + 8);                                            \
  } while (0)

template <bool BN, bool BM, bool GELU, bool RES, bool B16>
__global__ __launch_bounds__(256) void gemm_bf16_async(
    const unsigned short* __restrict__ A, int lda, long long bsA,
    const unsigned short* __restrict__ BT, long long bsB,
    void* __restrict__ Cout, int ldc, long long bsC,
    const float* __restrict__ bias_n, const float* __restrict__ bias_m,
    const float* __restrict__ residual, int K) {
  __shared__ __align__(16) unsigned short As[2][128 * 32];  // [m][k]
  __shared__ __align__(16) unsigned short Bs[2][128 * 32];  // [n][k]

  int z = blockIdx.z;
  A += (size_t)z * bsA;
  BT += (size_t)z * bsB;
  int m0 = blockIdx.y * 128;
  int n0 = blockIdx.x * 128;
  int tid = threadIdx.x;
  int lane = tid & 31;
  int w = tid >> 5;
  int wm = w >> 1;  // 0..3 -> 32-row strip
  int wn = w & 1;   // 0..1 -> 64-col strip

  v8f acc[2][4];
#pragma unroll
  for (int mi = 0; mi < 2; ++mi)
#pragma unroll
    for (int ni = 0; ni < 4; ++ni) acc[mi][ni] = v8f_zero();

  int am = tid >> 1, ac = (tid & 1) * 2;
  int bn = tid >> 1, bc = (tid & 1) * 2;
  const unsigned short* agp = A + (size_t)(m0 + am) * lda + ac * 8;
  const unsigned short* bgp = BT + (size_t)(n0 + bn) * K + bc * 8;
  unsigned aBase = lds_off(As) + (unsigned)(am * 64 + ac * 16);
  unsigned bBase = lds_off(Bs) + (unsigned)(bn * 64 + bc * 16);

  int nk = K / 32;
  GEMM_ISSUE(0, 0);
  for (int kt = 0; kt < nk; ++kt) {
    int buf = kt & 1;
    __syncthreads();                 // all waves done computing tile kt-1
    if (kt + 1 < nk) {
      GEMM_ISSUE(kt + 1, buf ^ 1);
      WAIT_ASYNC(0x4);               // tile kt's 4 requests have landed
    } else {
      WAIT_ASYNC(0x0);
    }
    __syncthreads();                 // tile kt visible to all waves

    const unsigned short* Ab = As[buf];
    const unsigned short* Bb = Bs[buf];
    v16bf a0 = lds_afrag(Ab, 32, wm * 32 + 0, 0, lane);
    v16bf a1 = lds_afrag(Ab, 32, wm * 32 + 16, 0, lane);
    v16bf b0 = lds_bfrag(Bb, 32, wn * 64 + 0, 0, lane);
    v16bf b1 = lds_bfrag(Bb, 32, wn * 64 + 16, 0, lane);
    v16bf b2 = lds_bfrag(Bb, 32, wn * 64 + 32, 0, lane);
    v16bf b3 = lds_bfrag(Bb, 32, wn * 64 + 48, 0, lane);
    acc[0][0] = wmma_bf16(a0, b0, acc[0][0]);
    acc[0][1] = wmma_bf16(a0, b1, acc[0][1]);
    acc[0][2] = wmma_bf16(a0, b2, acc[0][2]);
    acc[0][3] = wmma_bf16(a0, b3, acc[0][3]);
    acc[1][0] = wmma_bf16(a1, b0, acc[1][0]);
    acc[1][1] = wmma_bf16(a1, b1, acc[1][1]);
    acc[1][2] = wmma_bf16(a1, b2, acc[1][2]);
    acc[1][3] = wmma_bf16(a1, b3, acc[1][3]);
  }

  int hf = lane >> 4;
  int nl = lane & 15;
  float* Cf = (float*)Cout + (size_t)z * bsC;
  unsigned short* C16 = (unsigned short*)Cout + (size_t)z * bsC;
#pragma unroll
  for (int mi = 0; mi < 2; ++mi) {
#pragma unroll
    for (int ni = 0; ni < 4; ++ni) {
      int nn = n0 + wn * 64 + ni * 16 + nl;
      float bnv = BN ? bias_n[nn] : 0.0f;
#pragma unroll
      for (int r = 0; r < 8; ++r) {
        int mm = m0 + wm * 32 + mi * 16 + r + hf * 8;
        float v = acc[mi][ni][r] + bnv;
        if (BM) v += bias_m[mm];
        if (GELU) v = 0.5f * v * (1.0f + erff(v * 0.70710678118654752f));
        if (RES) v += residual[(size_t)mm * ldc + nn];
        if (B16) C16[(size_t)mm * ldc + nn] = f32_to_bf16_bits(v);
        else     Cf [(size_t)mm * ldc + nn] = v;
      }
    }
  }
}

// ---------------------------------------------------------------------------
// Fused flash attention, all-bf16 operands, 64-wide j-tiles, 2-deep pipeline.
// Q,K: (8,1024,512) bf16, head h at cols h*64.  VT: [(b*8+h)*64+dh][1024].
// grid = (1024/128, H, B), block 256 (8 waves x 16 query rows).
// ---------------------------------------------------------------------------
__global__ __launch_bounds__(256) void flash_attn_wmma(
    const unsigned short* __restrict__ Q, const unsigned short* __restrict__ Kb,
    const unsigned short* __restrict__ VT, unsigned short* __restrict__ O,
    float scale) {
  __shared__ __align__(16) unsigned short Ks[2][64 * 64];   // [j][d]
  __shared__ __align__(16) unsigned short Vs[2][64 * 64];   // [d][j]
  __shared__ __align__(16) unsigned short Ps[8][16 * 64];   // per-wave P [m][k]

  int tid = threadIdx.x;
  int lane = tid & 31;
  int w = tid >> 5;
  int hf = lane >> 4;
  int nl = lane & 15;

  size_t base = (size_t)blockIdx.z * 1024 * 512 + (size_t)blockIdx.y * 64;
  const unsigned short* q = Q + base;
  const unsigned short* k = Kb + base;
  const unsigned short* vt = VT + (size_t)(blockIdx.z * 8 + blockIdx.y) * 64 * 1024;
  unsigned short* o = O + base;

  int i0 = blockIdx.x * 128 + w * 16;
  v16bf qa0 = gld_afrag_bf16(q, 512, i0, 0, lane);
  v16bf qa1 = gld_afrag_bf16(q, 512, i0, 32, lane);

  v8f accO[4];
#pragma unroll
  for (int nf = 0; nf < 4; ++nf) accO[nf] = v8f_zero();
  float mrow[8], lrow[8];
#pragma unroll
  for (int r = 0; r < 8; ++r) { mrow[r] = -1e30f; lrow[r] = 0.0f; }

  // staging: K tile 64x64 and V tile 64x64, each 8KB = 2 chunks/thread
  int sr = tid >> 3, sc = tid & 7;           // row 0..31 (+32 for 2nd), chunk 0..7
  const unsigned short* kgp = k + (size_t)sr * 512 + sc * 8;
  const unsigned short* vgp = vt + (size_t)sr * 1024 + sc * 8;
  unsigned kBase = lds_off(Ks) + (unsigned)(sr * 128 + sc * 16);
  unsigned vBase = lds_off(Vs) + (unsigned)(sr * 128 + sc * 16);

#define ATTN_ISSUE(jt, buf)                                                   \
  do {                                                                        \
    unsigned _kb = kBase + (unsigned)(buf) * 16384u;                          \
    unsigned _vb = vBase + (unsigned)(buf) * 16384u;                          \
    const unsigned short* _kg = kgp + (size_t)(jt) * 64 * 512;                \
    const unsigned short* _vg = vgp + (jt) * 64;                              \
    ASYNC_B128(_kb, _kg);                                                     \
    ASYNC_B128(_kb + 4096u, _kg + (size_t)32 * 512);                          \
    ASYNC_B128(_vb, _vg);                                                     \
    ASYNC_B128(_vb + 4096u, _vg + (size_t)32 * 1024);                         \
  } while (0)

  ATTN_ISSUE(0, 0);
  for (int jt = 0; jt < 16; ++jt) {
    int buf = jt & 1;
    __syncthreads();
    if (jt + 1 < 16) {
      ATTN_ISSUE(jt + 1, buf ^ 1);
      WAIT_ASYNC(0x4);
    } else {
      WAIT_ASYNC(0x0);
    }
    __syncthreads();

    const unsigned short* Kt = Ks[buf];
    const unsigned short* Vb = Vs[buf];

    // S (16x64) = Q (16x64) @ K^T : 4 j-fragments x 2 contraction halves
    v8f S[4];
#pragma unroll
    for (int jf = 0; jf < 4; ++jf) {
      S[jf] = wmma_bf16(qa0, lds_bfrag(Kt, 64, jf * 16, 0, lane), v8f_zero());
      S[jf] = wmma_bf16(qa1, lds_bfrag(Kt, 64, jf * 16, 32, lane), S[jf]);
    }

    // online softmax per row (row r of the C-fragment = vgpr r across 16 lanes)
    unsigned short* Pw = Ps[w];
#pragma unroll
    for (int r = 0; r < 8; ++r) {
      float s0 = S[0][r] * scale, s1 = S[1][r] * scale;
      float s2 = S[2][r] * scale, s3 = S[3][r] * scale;
      float mx = fmaxf(fmaxf(s0, s1), fmaxf(s2, s3));
#pragma unroll
      for (int d = 1; d < 16; d <<= 1) mx = fmaxf(mx, __shfl_xor(mx, d, 32));
      float nm = fmaxf(mrow[r], mx);
      float p0 = __expf(s0 - nm), p1 = __expf(s1 - nm);
      float p2 = __expf(s2 - nm), p3 = __expf(s3 - nm);
      float corr = __expf(mrow[r] - nm);
      mrow[r] = nm;
      float rs = (p0 + p1) + (p2 + p3);
#pragma unroll
      for (int d = 1; d < 16; d <<= 1) rs += __shfl_xor(rs, d, 32);
      lrow[r] = lrow[r] * corr + rs;
#pragma unroll
      for (int nf = 0; nf < 4; ++nf) accO[nf][r] *= corr;
      int mrw = r + hf * 8;
      Pw[mrw * 64 + nl]      = f32_to_bf16_bits(p0);
      Pw[mrw * 64 + 16 + nl] = f32_to_bf16_bits(p1);
      Pw[mrw * 64 + 32 + nl] = f32_to_bf16_bits(p2);
      Pw[mrw * 64 + 48 + nl] = f32_to_bf16_bits(p3);
    }

    // O += P (16x64) @ V (64x64)
    v16bf pa0 = lds_afrag(Pw, 64, 0, 0, lane);
    v16bf pa1 = lds_afrag(Pw, 64, 0, 32, lane);
#pragma unroll
    for (int nf = 0; nf < 4; ++nf) {
      accO[nf] = wmma_bf16(pa0, lds_bfrag(Vb, 64, nf * 16, 0, lane), accO[nf]);
      accO[nf] = wmma_bf16(pa1, lds_bfrag(Vb, 64, nf * 16, 32, lane), accO[nf]);
    }
  }
#undef ATTN_ISSUE

#pragma unroll
  for (int nf = 0; nf < 4; ++nf) {
    int dh = nf * 16 + nl;
#pragma unroll
    for (int r = 0; r < 8; ++r) {
      int row = i0 + r + hf * 8;
      o[(size_t)row * 512 + dh] = f32_to_bf16_bits(accO[nf][r] / lrow[r]);
    }
  }
}

// ---------------------------------------------------------------------------
// Host orchestration
// ---------------------------------------------------------------------------
extern "C" void kernel_launch(void* const* d_in, const int* in_sizes, int n_in,
                              void* d_out, int out_size, void* d_ws, size_t ws_size,
                              hipStream_t stream) {
  const float* in_x  = (const float*)d_in[0];
  const float* in_m  = (const float*)d_in[1];
  const float* in_n  = (const float*)d_in[2];
  const float* lin_w = (const float*)d_in[3];
  const float* lin_b = (const float*)d_in[4];
  const float* Wq    = (const float*)d_in[5];
  const float* Wk    = (const float*)d_in[6];
  const float* Wv    = (const float*)d_in[7];
  const float* Wo    = (const float*)d_in[8];
  const float* bo    = (const float*)d_in[9];
  const float* ln1_g = (const float*)d_in[10];
  const float* ln1_b = (const float*)d_in[11];
  const float* W1    = (const float*)d_in[12];
  const float* b1    = (const float*)d_in[13];
  const float* W2    = (const float*)d_in[14];
  const float* b2    = (const float*)d_in[15];
  const float* ln2_g = (const float*)d_in[16];
  const float* ln2_b = (const float*)d_in[17];

  const size_t MB = 1024 * 1024;
  char* wsb = (char*)d_ws;
  float*          x    = (float*)(wsb);                      // 16 MB f32 residual
  unsigned short* hbuf = (unsigned short*)(wsb + 16 * MB);   //  8 MB LN / attn out
  unsigned short* qb   = (unsigned short*)(wsb + 24 * MB);
  unsigned short* kb   = (unsigned short*)(wsb + 32 * MB);
  unsigned short* vb   = (unsigned short*)(wsb + 40 * MB);
  unsigned short* vT   = (unsigned short*)(wsb + 48 * MB);
  unsigned short* mlph = (unsigned short*)(wsb + 56 * MB);   // 32 MB
  unsigned short* linT = (unsigned short*)(wsb + 88 * MB);
  unsigned short* inxT = (unsigned short*)(wsb + 89 * MB);
  unsigned short* WqT  = (unsigned short*)(wsb + 91 * MB);
  unsigned short* WkT  = (unsigned short*)(wsb + 93 * MB);
  unsigned short* WvT  = (unsigned short*)(wsb + 95 * MB);
  unsigned short* WoT  = (unsigned short*)(wsb + 97 * MB);
  unsigned short* W1T  = (unsigned short*)(wsb + 99 * MB);
  unsigned short* W2T  = (unsigned short*)(wsb + 107 * MB);

  dim3 blk(256);
  const int ROWS = 8 * 1024;
  const float scale = 0.04419417382415922f;   // DIM^-0.5 (dim, not dim_head)

  // ---- one-time layout pass: bf16 + B operands as [N][K] ----
  transpose_f32_bf16<<<dim3(32, 8, 1), blk, 0, stream>>>(lin_w, linT, 256, 1024, 0, 0);
  transpose_f32_bf16<<<dim3(16, 8, 8), blk, 0, stream>>>(in_x, inxT, 256, 512,
                                                         256LL * 512, 256LL * 512);
  transpose_f32_bf16<<<dim3(16, 16, 4), blk, 0, stream>>>(Wq, WqT, 512, 512,
                                                          512LL * 512, 512LL * 512);
  transpose_f32_bf16<<<dim3(16, 16, 4), blk, 0, stream>>>(Wk, WkT, 512, 512,
                                                          512LL * 512, 512LL * 512);
  transpose_f32_bf16<<<dim3(16, 16, 4), blk, 0, stream>>>(Wv, WvT, 512, 512,
                                                          512LL * 512, 512LL * 512);
  transpose_f32_bf16<<<dim3(16, 16, 4), blk, 0, stream>>>(Wo, WoT, 512, 512,
                                                          512LL * 512, 512LL * 512);
  transpose_f32_bf16<<<dim3(64, 16, 4), blk, 0, stream>>>(W1, W1T, 512, 2048,
                                                          512LL * 2048, 512LL * 2048);
  transpose_f32_bf16<<<dim3(16, 64, 4), blk, 0, stream>>>(W2, W2T, 2048, 512,
                                                          2048LL * 512, 2048LL * 512);

  // ---- x[b,l,d] = sum_s lin_w[s,l] in_x[b,s,d] + lin_b[l]
  gemm_bf16_async<false, true, false, false, false>
      <<<dim3(4, 8, 8), blk, 0, stream>>>(
          linT, 256, 0LL, inxT, 512LL * 256, x, 512, 1024LL * 512,
          nullptr, lin_b, nullptr, 256);

  for (int i = 0; i < 4; ++i) {
    const unsigned short* WqTi = WqT + (size_t)i * 512 * 512;
    const unsigned short* WkTi = WkT + (size_t)i * 512 * 512;
    const unsigned short* WvTi = WvT + (size_t)i * 512 * 512;
    const unsigned short* WoTi = WoT + (size_t)i * 512 * 512;
    const unsigned short* W1Ti = W1T + (size_t)i * 2048 * 512;
    const unsigned short* W2Ti = W2T + (size_t)i * 512 * 2048;

    // q = LN1(x) @ Wq
    layernorm_512_bf16<<<ROWS, blk, 0, stream>>>(x, ln1_g + i * 512, ln1_b + i * 512, hbuf);
    gemm_bf16_async<false, false, false, false, true>
        <<<dim3(4, 64, 1), blk, 0, stream>>>(
            hbuf, 512, 0LL, WqTi, 0LL, qb, 512, 0LL, nullptr, nullptr, nullptr, 512);
    // k = LN1(m) @ Wk
    layernorm_512_bf16<<<ROWS, blk, 0, stream>>>(in_m, ln1_g + i * 512, ln1_b + i * 512, hbuf);
    gemm_bf16_async<false, false, false, false, true>
        <<<dim3(4, 64, 1), blk, 0, stream>>>(
            hbuf, 512, 0LL, WkTi, 0LL, kb, 512, 0LL, nullptr, nullptr, nullptr, 512);
    // v = LN1(n) @ Wv, then per-head re-tile
    layernorm_512_bf16<<<ROWS, blk, 0, stream>>>(in_n, ln1_g + i * 512, ln1_b + i * 512, hbuf);
    gemm_bf16_async<false, false, false, false, true>
        <<<dim3(4, 64, 1), blk, 0, stream>>>(
            hbuf, 512, 0LL, WvTi, 0LL, vb, 512, 0LL, nullptr, nullptr, nullptr, 512);
    transpose_v_heads<<<dim3(32, 8, 8), blk, 0, stream>>>(vb, vT);

    // attention -> hbuf (bf16)
    flash_attn_wmma<<<dim3(8, 8, 8), blk, 0, stream>>>(qb, kb, vT, hbuf, scale);

    // x = x + attn @ Wo + bo   (f32, in-place residual)
    gemm_bf16_async<true, false, false, true, false>
        <<<dim3(4, 64, 1), blk, 0, stream>>>(
            hbuf, 512, 0LL, WoTi, 0LL, x, 512, 0LL, bo + i * 512, nullptr, x, 512);

    // h = gelu(LN2(x) @ W1 + b1)  (bf16)
    layernorm_512_bf16<<<ROWS, blk, 0, stream>>>(x, ln2_g + i * 512, ln2_b + i * 512, hbuf);
    gemm_bf16_async<true, false, true, false, true>
        <<<dim3(16, 64, 1), blk, 0, stream>>>(
            hbuf, 512, 0LL, W1Ti, 0LL, mlph, 2048, 0LL, b1 + i * 2048, nullptr, nullptr, 512);

    // x = x + h @ W2 + b2   (last layer -> d_out)
    float* dst = (i == 3) ? (float*)d_out : x;
    gemm_bf16_async<true, false, false, true, false>
        <<<dim3(4, 64, 1), blk, 0, stream>>>(
            mlph, 2048, 0LL, W2Ti, 0LL, dst, 512, 0LL, b2 + i * 512, nullptr, x, 2048);
  }
}
